// EmbeddingLSTMCell_28845000360212
// MI455X (gfx1250) — matrix operand order repaired and checked by reading, not compiled, not run on hardware
//
#include <hip/hip_runtime.h>

// LSTM cell fused kernel for gfx1250 (MI455X):
//   Pre-pass: f32 -> bf16 conversion of weights (transposed) and [x|h] activations into d_ws.
//   Main:     z = cat @ [Wf|Wi|Wc|Wo] via v_wmma_f32_16x16x32_bf16, LDS fed by a
//             3-stage GLOBAL_LOAD_ASYNC_TO_LDS_B128 pipeline (s_wait_asynccnt 0x5),
//             branch-free steady-state loop (last two iterations peeled),
//             fused sigmoid/tanh epilogue.
//
// B=8192, IN=2048, HID=2048, K = IN+HID = 4096.

#define BATCH 8192
#define INDIM 2048
#define HIDDIM 2048
#define KTOT (INDIM + HIDDIM)

#define BM 64
#define BN 64
#define BK 32
#define NK (KTOT / BK)
#define LDST 40                          // LDS row stride in bf16 elements (80 B)
#define STG_ELEMS ((BM + 4 * BN) * LDST) // 12800 bf16 per stage
#define STG_BYTES (STG_ELEMS * 2)        // 25600 B per stage
#define NSTAGE 3

typedef __attribute__((ext_vector_type(16))) __bf16 v16bf;
typedef __attribute__((ext_vector_type(8)))  __bf16 v8bf;
typedef __attribute__((ext_vector_type(8)))  float  v8f;

__device__ __forceinline__ float fsigmoid(float x) {
    return 1.0f / (1.0f + __expf(-x));
}
__device__ __forceinline__ float ftanh(float x) {
    float e2 = __expf(2.0f * x);
    return (e2 - 1.0f) / (e2 + 1.0f);
}

// Async copy 16 bytes/lane from global to LDS (tracked by ASYNCcnt).
__device__ __forceinline__ void async_b128(unsigned lds_byte_off, const void* gaddr) {
    asm volatile("global_load_async_to_lds_b128 %0, %1, off"
                 :: "v"(lds_byte_off), "v"((unsigned long long)(size_t)gaddr)
                 : "memory");
}
// Allow the most recent stage's 5 copies to remain in flight.
__device__ __forceinline__ void wait_async5() {
    asm volatile("s_wait_asynccnt 0x5" ::: "memory");
}
__device__ __forceinline__ void wait_async0() {
    asm volatile("s_wait_asynccnt 0x0" ::: "memory");
}

// ---------------- Pre-pass 1: weights f32 [k][n] -> bf16 transposed Wt[g][n][k] ----------------
__global__ __launch_bounds__(256)
void weight_convert(const float* __restrict__ Wf, const float* __restrict__ Wi,
                    const float* __restrict__ Wc, const float* __restrict__ Wo,
                    __bf16* __restrict__ Wt)
{
    __shared__ __attribute__((aligned(32))) __bf16 tile[64 * 72];  // [n][k], pad 72
    const float* Wg[4] = {Wf, Wi, Wc, Wo};
    const float* W = Wg[blockIdx.z];
    const int kb = blockIdx.x * 64;
    const int nb = blockIdx.y * 64;
    const int tid = threadIdx.x;

#pragma unroll
    for (int i = 0; i < 4; ++i) {
        int idx = tid + i * 256;          // 0..1023 -> 64k x 16 float4
        int k  = idx >> 4;                // 0..63
        int n4 = (idx & 15) * 4;          // 0..60
        float4 v = *(const float4*)(W + (size_t)(kb + k) * HIDDIM + nb + n4);
        tile[(n4 + 0) * 72 + k] = (__bf16)v.x;
        tile[(n4 + 1) * 72 + k] = (__bf16)v.y;
        tile[(n4 + 2) * 72 + k] = (__bf16)v.z;
        tile[(n4 + 3) * 72 + k] = (__bf16)v.w;
    }
    __syncthreads();

    __bf16* dst = Wt + ((size_t)blockIdx.z * HIDDIM + nb) * KTOT + kb;
#pragma unroll
    for (int i = 0; i < 2; ++i) {
        int idx = tid + i * 256;          // 0..511
        int n  = idx >> 3;                // 0..63
        int kc = (idx & 7) * 8;           // 0..56
        v8bf v = *(const v8bf*)&tile[n * 72 + kc];
        *(v8bf*)(dst + (size_t)n * KTOT + kc) = v;
    }
}

// ---------------- Pre-pass 2: activations cat=[x|h] f32 -> bf16 Abf[m][KTOT] ----------------
__global__ __launch_bounds__(256)
void act_convert(const float* __restrict__ x, const float* __restrict__ h,
                 __bf16* __restrict__ A)
{
    size_t e = ((size_t)blockIdx.x * 256 + threadIdx.x) * 8;
    size_t m = e >> 12;                  // /4096
    int k = (int)(e & 4095);
    const float* src = (k < INDIM) ? (x + m * INDIM + k)
                                   : (h + m * HIDDIM + (k - INDIM));
    float4 a = ((const float4*)src)[0];
    float4 b = ((const float4*)src)[1];
    v8bf o;
    o[0] = (__bf16)a.x; o[1] = (__bf16)a.y; o[2] = (__bf16)a.z; o[3] = (__bf16)a.w;
    o[4] = (__bf16)b.x; o[5] = (__bf16)b.y; o[6] = (__bf16)b.z; o[7] = (__bf16)b.w;
    *(v8bf*)(A + e) = o;
}

// One K-step of the 4-gate WMMA tile from LDS stage `st`.
__device__ __forceinline__ void compute_stage(const __bf16* st, v8f acc[4][2],
                                              int arow, int akb, int bkb,
                                              int wn, int lane)
{
    v8bf alo = *(const v8bf*)&st[arow * LDST + akb];
    v8bf ahi = *(const v8bf*)&st[arow * LDST + akb + 16];
    v16bf afrag = __builtin_shufflevector(alo, ahi,
                      0, 1, 2, 3, 4, 5, 6, 7, 8, 9, 10, 11, 12, 13, 14, 15);

#pragma unroll
    for (int g = 0; g < 4; ++g) {
        const __bf16* ls = st + (BM + g * BN) * LDST;
#pragma unroll
        for (int t = 0; t < 2; ++t) {
            int brow = wn + t * 16 + (lane & 15);
            v8bf blo = *(const v8bf*)&ls[brow * LDST + bkb];
            v8bf bhi = *(const v8bf*)&ls[brow * LDST + bkb + 16];
            v16bf bfrag = __builtin_shufflevector(blo, bhi,
                              0, 1, 2, 3, 4, 5, 6, 7, 8, 9, 10, 11, 12, 13, 14, 15);
            acc[g][t] = __builtin_amdgcn_wmma_f32_16x16x32_bf16(
                false, afrag, false, bfrag, (short)0, acc[g][t], false, false);
        }
    }
}

// ---------------- Main fused WMMA kernel ----------------
__global__ __launch_bounds__(256)
void lstm_fused_wmma(const __bf16* __restrict__ Abf, const __bf16* __restrict__ Wt,
                     const float* __restrict__ c_prev,
                     const float* __restrict__ bF, const float* __restrict__ bI,
                     const float* __restrict__ bC, const float* __restrict__ bO,
                     float* __restrict__ out)
{
    __shared__ __attribute__((aligned(128))) __bf16 smem[NSTAGE * STG_ELEMS];

    const int tid    = threadIdx.x;
    const int lane   = tid & 31;
    const int wave   = tid >> 5;
    const int blockN = blockIdx.x * BN;
    const int blockM = blockIdx.y * BM;

    const int wm = (wave >> 1) * 16;   // wave row offset in tile
    const int wn = (wave & 1) * 32;    // wave col offset in tile

    v8f acc[4][2];
    const v8f vzero = {0.f, 0.f, 0.f, 0.f, 0.f, 0.f, 0.f, 0.f};
#pragma unroll
    for (int g = 0; g < 4; ++g) { acc[g][0] = vzero; acc[g][1] = vzero; }

    // Fragment addressing (ISA 7.12.2 16-bit layouts)
    const int arow = wm + (lane & 15);
    const int akb  = (lane >> 4) * 8;
    const int bkb  = (lane >> 4) * 8;

    // Async-copy coordinates: each lane moves 16 B per tile (256 lanes x 16 B = 4 KB tile)
    const int cp_r  = tid >> 2;          // 0..63 (row of A tile / n of B tile)
    const int cp_kc = (tid & 3) * 8;     // bf16 k-chunk base: 0,8,16,24

    const unsigned smem_base = (unsigned)(size_t)&smem[0];
    const unsigned aLds = (unsigned)(cp_r * (LDST * 2) + cp_kc * 2);
    unsigned bLds[4];
    const __bf16* aG = Abf + (size_t)(blockM + cp_r) * KTOT + cp_kc;
    const __bf16* bG[4];
#pragma unroll
    for (int g = 0; g < 4; ++g) {
        bLds[g] = (unsigned)((BM + g * BN + cp_r) * (LDST * 2) + cp_kc * 2);
        bG[g] = Wt + ((size_t)g * HIDDIM + blockN + cp_r) * KTOT + cp_kc;
    }

    // Prologue: fill stage 0 and stage 1
    {
        async_b128(smem_base + aLds, aG);
#pragma unroll
        for (int g = 0; g < 4; ++g) async_b128(smem_base + bLds[g], bG[g]);
        unsigned sb = smem_base + STG_BYTES;
        async_b128(sb + aLds, aG + BK);
#pragma unroll
        for (int g = 0; g < 4; ++g) async_b128(sb + bLds[g], bG[g] + BK);
    }

    // Rotating stage byte-offsets: off0 = compute, off1 = in flight, off2 = issue target
    unsigned off0 = 0, off1 = STG_BYTES, off2 = 2 * STG_BYTES;

    // Steady state: branch-free (stage kk+2 always exists, newest 5 copies may stay in flight)
#pragma unroll 1
    for (int kk = 0; kk < NK - 2; ++kk) {
        wait_async5();
        __syncthreads();

        unsigned sb = smem_base + off2;
        int koff = (kk + 2) * BK;
        async_b128(sb + aLds, aG + koff);
#pragma unroll
        for (int g = 0; g < 4; ++g) async_b128(sb + bLds[g], bG[g] + koff);

        compute_stage((const __bf16*)((const char*)smem + off0), acc,
                      arow, akb, bkb, wn, lane);

        unsigned tmp = off0; off0 = off1; off1 = off2; off2 = tmp;
    }

    // Peeled iteration NK-2: stage NK-1 still in flight, nothing new to issue.
    wait_async5();
    __syncthreads();
    compute_stage((const __bf16*)((const char*)smem + off0), acc,
                  arow, akb, bkb, wn, lane);
    { unsigned tmp = off0; off0 = off1; off1 = off2; off2 = tmp; }

    // Peeled iteration NK-1: full drain.
    wait_async0();
    __syncthreads();
    compute_stage((const __bf16*)((const char*)smem + off0), acc,
                  arow, akb, bkb, wn, lane);

    // ---- fused LSTM epilogue ----
    const size_t OUTOFF = (size_t)BATCH * HIDDIM;   // c_t follows h_t
#pragma unroll
    for (int t = 0; t < 2; ++t) {
        int ng = blockN + wn + t * 16 + (lane & 15);
        float vbf = bF[ng];
        float vbi = bI[ng];
        float vbc = bC[ng];
        float vbo = bO[ng];
#pragma unroll
        for (int v = 0; v < 8; ++v) {
            int mg = blockM + wm + v + ((lane >> 4) * 8);
            float zf = acc[0][t][v] + vbf;
            float zi = acc[1][t][v] + vbi;
            float zc = acc[2][t][v] + vbc;
            float zo = acc[3][t][v] + vbo;
            float fg = fsigmoid(zf);
            float ig = fsigmoid(zi);
            float cc = ftanh(zc);
            float og = fsigmoid(zo);
            float cp = c_prev[(size_t)mg * HIDDIM + ng];
            float ct = cp * fg + ig * cc;
            float ht = og * ftanh(ct);
            out[(size_t)mg * HIDDIM + ng]          = ht;
            out[OUTOFF + (size_t)mg * HIDDIM + ng] = ct;
        }
    }
}

extern "C" void kernel_launch(void* const* d_in, const int* in_sizes, int n_in,
                              void* d_out, int out_size, void* d_ws, size_t ws_size,
                              hipStream_t stream) {
    const float* x      = (const float*)d_in[0];
    const float* h_prev = (const float*)d_in[1];
    const float* c_prev = (const float*)d_in[2];
    // d_in[3] = embedding_vec (ignored in vanilla mode)
    const float* Wf = (const float*)d_in[4];
    const float* Wi = (const float*)d_in[5];
    const float* Wc = (const float*)d_in[6];
    const float* Wo = (const float*)d_in[7];
    const float* bF = (const float*)d_in[8];
    const float* bI = (const float*)d_in[9];
    const float* bC = (const float*)d_in[10];
    const float* bO = (const float*)d_in[11];
    float* out = (float*)d_out;

    // Workspace layout: Abf (64 MB bf16) then Wt (64 MB bf16, transposed per gate)
    __bf16* Abf = (__bf16*)d_ws;
    __bf16* Wt  = (__bf16*)((char*)d_ws + (size_t)BATCH * KTOT * sizeof(__bf16));

    {   // activations: 8192*4096/8 elems per thread-chunk / 256 threads
        dim3 grid((unsigned)(((size_t)BATCH * KTOT / 8) / 256));
        act_convert<<<grid, 256, 0, stream>>>(x, h_prev, Abf);
    }
    {   // weights: 64x64 transpose tiles
        dim3 grid(KTOT / 64, HIDDIM / 64, 4);
        weight_convert<<<grid, 256, 0, stream>>>(Wf, Wi, Wc, Wo, Wt);
    }
    {
        dim3 grid(HIDDIM / BN, BATCH / BM);   // (32, 128)
        lstm_fused_wmma<<<grid, 256, 0, stream>>>(
            Abf, Wt, c_prev, bF, bI, bC, bO, out);
    }
}
